// Attention_48421461295734
// MI455X (gfx1250) — matrix operand order repaired and checked by reading, not compiled, not run on hardware
//
#include <hip/hip_runtime.h>
#include <hip/hip_bf16.h>

// ---------------------------------------------------------------------------
// RoPE'd strictly-causal linear attention (no softmax) for MI455X / gfx1250.
//   Q: (1,4,2048,8192) f32   V: (1,1,2048,256) f32   Out: (1,4,2048,256) f32
// QK^T with K=8192 dominates (~275 GFLOP) -> bf16 WMMA, f32 accumulate.
// QR in bf16 = 128 MB: L2-resident (192 MB), so GEMM operands stream from L2.
// v3: 32q x 64k register tiles -> every B fragment feeds 2 WMMAs:
//     12 b128 loads per 8 WMMAs = 21.3 FLOP/byte of L2 traffic (vs 12.8 in v2).
//     4-way split-K over keys for balanced wave32 scheduling + deterministic
//     fixed-order reduction (no float atomics).
// ---------------------------------------------------------------------------

#define NH    4
#define TDIM  2048
#define NDIM  8192
#define DDIM  256
#define KSPLIT  4
#define KCHUNK  512           // TDIM / KSPLIT keys per task
#define QT32    (TDIM / 32)   // 64 query tiles of 32 rows

typedef __attribute__((ext_vector_type(16))) __bf16 v16bf;
typedef __attribute__((ext_vector_type(8)))  __bf16 v8bf;   // 16 B -> one b128
typedef __attribute__((ext_vector_type(8)))  float  v8f;

union FragBF16 { v16bf v; v8bf h[2]; };

// ---------------------------------------------------------------------------
// Kernel 1: RoPE(Q) -> bf16.  freq = 2^{-16*(2p)/N} / (2pi); pairwise rotate.
// ---------------------------------------------------------------------------
__global__ void rope_bf16_kernel(const float* __restrict__ Q,
                                 __bf16* __restrict__ QR, long total_pairs) {
  long i = (long)blockIdx.x * blockDim.x + threadIdx.x;
  if (i >= total_pairs) return;
  int  p   = (int)(i % (NDIM / 2));
  long row = i / (NDIM / 2);              // row = h*T + t
  int  t   = (int)(row % TDIM);
  float freq  = exp2f(-16.0f * (float)(2 * p) / (float)NDIM) * 0.15915494309189535f;
  float phase = (float)t * freq;
  float frac  = phase - floorf(phase);
  float ang   = frac * 6.283185307179586f;
  float c = __cosf(ang), s = __sinf(ang);
  long base = row * (long)NDIM + 2 * p;
  float q0 = Q[base], q1 = Q[base + 1];
  QR[base]     = (__bf16)(q0 * c - q1 * s);
  QR[base + 1] = (__bf16)(q1 * c + q0 * s);
}

// ---------------------------------------------------------------------------
// Kernel 2: V (t,d) f32 -> VT (d,t) bf16 so PV B-fragments read contiguous K.
// ---------------------------------------------------------------------------
__global__ void vt_bf16_kernel(const float* __restrict__ V,
                               __bf16* __restrict__ VT) {
  int i = blockIdx.x * blockDim.x + threadIdx.x;
  if (i >= TDIM * DDIM) return;
  int t = i / DDIM, d = i % DDIM;
  VT[(long)d * TDIM + t] = (__bf16)V[i];
}

// ---------------------------------------------------------------------------
// Kernel 3: one wave per task = (head, 32-query tile, 512-key chunk).
// For each 64-key tile in the chunk:
//   S(32x64) += A(16x32)x2 x B(32x16)x4 over K=8192  (2048 wmma, 1.5 ld/wmma)
//   strict-causal mask -> bf16 -> LDS -> A-frags, then
//   Opart(32x256) += S(32x64) x VT(64x256)           (64 wmma)
// Fragment layouts per CDNA5 ISA 7.12.2 (16-bit, wave32):
//   A: lane = row, half = lane>>4; regs 0-3: K=half*8+0..7, regs 4-7: +16
//   B: lane = col, half = lane>>4; regs 0-3: K=half*8+0..7, regs 4-7: +16 (same as A)
// Partial output always written (zeros for empty chunks) -> deterministic.
// ---------------------------------------------------------------------------
__launch_bounds__(32)
__global__ void attn_partial_kernel(const __bf16* __restrict__ QR,
                                    const __bf16* __restrict__ VT,
                                    float* __restrict__ PART) {
  __shared__ __align__(16) __bf16 sS[32 * 64];

  const int lane = threadIdx.x;
  const int half = lane >> 4;
  const int l16  = lane & 15;
  const int task   = blockIdx.x;            // ((head*QT32 + qtile)*4) + kchunk
  const int kchunk = task & (KSPLIT - 1);
  const int qtile  = (task >> 2) & (QT32 - 1);
  const int head   = task >> 8;             // 2 (kchunk) + 6 (qtile) bits
  const int qbase  = qtile << 5;

  const int kbeg = kchunk * KCHUNK;
  const int klim = qbase + 32;              // strict-causal upper bound
  const int kend = (kbeg + KCHUNK < klim) ? (kbeg + KCHUNK) : klim;

  v8f oacc[2][16];
  #pragma unroll
  for (int mt = 0; mt < 2; ++mt)
    #pragma unroll
    for (int d = 0; d < 16; ++d) oacc[mt][d] = (v8f){};

  if (kbeg < kend) {
    const __bf16* Qh = QR + (long)head * TDIM * NDIM;
    const __bf16* arow[2];
    #pragma unroll
    for (int mt = 0; mt < 2; ++mt)
      arow[mt] = Qh + (long)(qbase + 16 * mt + l16) * NDIM;

    for (int kb = kbeg; kb < kend; kb += 64) {
      v8f s[2][4];
      #pragma unroll
      for (int mt = 0; mt < 2; ++mt)
        #pragma unroll
        for (int j = 0; j < 4; ++j) s[mt][j] = (v8f){};
      const __bf16* brow[4];
      #pragma unroll
      for (int j = 0; j < 4; ++j)
        brow[j] = Qh + (long)(kb + 16 * j + l16) * NDIM;

      for (int k0 = 0; k0 < NDIM; k0 += 32) {
        FragBF16 a0, a1;
        a0.h[0] = *(const v8bf*)(arow[0] + k0 + half * 8);
        a0.h[1] = *(const v8bf*)(arow[0] + k0 + 16 + half * 8);
        a1.h[0] = *(const v8bf*)(arow[1] + k0 + half * 8);
        a1.h[1] = *(const v8bf*)(arow[1] + k0 + 16 + half * 8);
        #pragma unroll
        for (int j = 0; j < 4; ++j) {
          FragBF16 b;                        // shared by both query row-tiles
          b.h[0] = *(const v8bf*)(brow[j] + k0 + half * 8);
          b.h[1] = *(const v8bf*)(brow[j] + k0 + 16 + half * 8);
          s[0][j] = __builtin_amdgcn_wmma_f32_16x16x32_bf16(false, a0.v, false, b.v,
                                                            (short)0, s[0][j], false, false);
          s[1][j] = __builtin_amdgcn_wmma_f32_16x16x32_bf16(false, a1.v, false, b.v,
                                                            (short)0, s[1][j], false, false);
        }
      }

      // Strict causal mask (key < query), f32 -> bf16, stage 32x64 tile in LDS.
      // C-tile layout: lane col = l16, row = r + 8*half for reg r.
      #pragma unroll
      for (int mt = 0; mt < 2; ++mt) {
        #pragma unroll
        for (int j = 0; j < 4; ++j) {
          #pragma unroll
          for (int r = 0; r < 8; ++r) {
            int m    = 16 * mt + r + half * 8;
            int qrow = qbase + m;
            int col  = 16 * j + l16;
            sS[m * 64 + col] = (kb + col < qrow) ? (__bf16)s[mt][j][r] : (__bf16)0.0f;
          }
        }
      }
      __syncthreads();                       // single-wave WG: orders LDS st->ld

      FragBF16 a2[2][2];                     // S row-tiles as A-frags (K 0..31, 32..63)
      #pragma unroll
      for (int mt = 0; mt < 2; ++mt) {
        const __bf16* srow = &sS[(16 * mt + l16) * 64];
        a2[mt][0].h[0] = *(const v8bf*)(srow + half * 8);
        a2[mt][0].h[1] = *(const v8bf*)(srow + 16 + half * 8);
        a2[mt][1].h[0] = *(const v8bf*)(srow + 32 + half * 8);
        a2[mt][1].h[1] = *(const v8bf*)(srow + 48 + half * 8);
      }
      __syncthreads();                       // protect sS before next key tile

      // Opart += S x V : B[k][n] = VT[dim][kb+k] (row-major VT rows)
      #pragma unroll
      for (int dt = 0; dt < 16; ++dt) {
        const __bf16* vrow = VT + (long)(dt * 16 + l16) * TDIM + kb;
        FragBF16 b0, b1;
        b0.h[0] = *(const v8bf*)(vrow + half * 8);
        b0.h[1] = *(const v8bf*)(vrow + 16 + half * 8);
        b1.h[0] = *(const v8bf*)(vrow + 32 + half * 8);
        b1.h[1] = *(const v8bf*)(vrow + 48 + half * 8);
        #pragma unroll
        for (int mt = 0; mt < 2; ++mt) {
          oacc[mt][dt] = __builtin_amdgcn_wmma_f32_16x16x32_bf16(
              false, a2[mt][0].v, false, b0.v, (short)0, oacc[mt][dt], false, false);
          oacc[mt][dt] = __builtin_amdgcn_wmma_f32_16x16x32_bf16(
              false, a2[mt][1].v, false, b1.v, (short)0, oacc[mt][dt], false, false);
        }
      }
    }
  }

  // Write 32x256 partial tile (zeros if chunk empty -> no pre-clear needed).
  float* part = PART + (size_t)task * (32 * DDIM);
  #pragma unroll
  for (int mt = 0; mt < 2; ++mt) {
    #pragma unroll
    for (int dt = 0; dt < 16; ++dt) {
      #pragma unroll
      for (int r = 0; r < 8; ++r) {
        int m = 16 * mt + r + half * 8;
        part[m * DDIM + dt * 16 + l16] = oacc[mt][dt][r];
      }
    }
  }
}

// ---------------------------------------------------------------------------
// Kernel 4: deterministic fixed-order split-K reduction: Out = sum of 4 parts.
// ---------------------------------------------------------------------------
__global__ void reduce_kernel(const float* __restrict__ PART,
                              float* __restrict__ Out) {
  int i = blockIdx.x * blockDim.x + threadIdx.x;   // over NH*T*D
  if (i >= NH * TDIM * DDIM) return;
  int d = i % DDIM;
  int q = (i / DDIM) % TDIM;
  int h = i / (DDIM * TDIM);
  int qtile = q >> 5, row = q & 31;
  const int TILE = 32 * DDIM;                       // 8192 floats per partial
  size_t base = (size_t)((h * QT32 + qtile) * KSPLIT) * TILE + row * DDIM + d;
  float sum = ((PART[base] + PART[base + TILE]) +
               (PART[base + 2 * TILE] + PART[base + 3 * TILE]));
  Out[i] = sum;
}

// ---------------------------------------------------------------------------
extern "C" void kernel_launch(void* const* d_in, const int* in_sizes, int n_in,
                              void* d_out, int out_size, void* d_ws, size_t ws_size,
                              hipStream_t stream) {
  const float* Q = (const float*)d_in[0];   // (1,4,2048,8192) f32
  const float* V = (const float*)d_in[1];   // (1,1,2048,256)  f32
  float* Out = (float*)d_out;               // (1,4,2048,256)  f32

  __bf16* QR  = (__bf16*)d_ws;                                  // 128 MB
  __bf16* VT  = QR + (size_t)NH * TDIM * NDIM;                  // + 1 MB
  float*  PRT = (float*)(VT + (size_t)DDIM * TDIM);             // + 32 MB

  long pairs = (long)NH * TDIM * (NDIM / 2);
  rope_bf16_kernel<<<(unsigned)((pairs + 255) / 256), 256, 0, stream>>>(Q, QR, pairs);

  int vtn = TDIM * DDIM;
  vt_bf16_kernel<<<(vtn + 255) / 256, 256, 0, stream>>>(V, VT);

  int ntasks = NH * QT32 * KSPLIT;          // 1024 single-wave workgroups
  attn_partial_kernel<<<ntasks, 32, 0, stream>>>(QR, VT, PRT);

  int nout = NH * TDIM * DDIM;
  reduce_kernel<<<(nout + 255) / 256, 256, 0, stream>>>(PRT, Out);
}
